// SmilesVAE_79766132621864
// MI455X (gfx1250) — matrix operand order repaired and checked by reading, not compiled, not run on hardware
//
#include <hip/hip_runtime.h>
#include <hip/hip_bf16.h>
#include <math.h>

// Problem constants (SmilesVAE reference): V,E,H,L,B,T
#define VV 50
#define EE 128
#define HH 128
#define LL_ 128
#define BB 4096
#define TT 100

typedef __attribute__((ext_vector_type(16))) __bf16 bf16x16;
typedef __attribute__((ext_vector_type(8)))  float  f32x8;

union ABu { bf16x16 v; uint4 q[2]; };

// ---- WMMA fragment loaders (ISA 7.12.2 layouts, wave32) -------------------
// A (16x32 bf16, MxK): lane L<16 holds row M=L, K=[k0..k0+8) in elems 0-7 and
// K=[k0+16..k0+24) in elems 8-15; lanes 16-31 same rows, K shifted by +8.
__device__ __forceinline__ bf16x16 loadA_row(const __bf16* base, int ldK,
                                             int row, int kBase, int half) {
  const __bf16* p = base + (size_t)row * ldK + kBase + half * 8;
  ABu t;
  t.q[0] = *(const uint4*)(p);
  t.q[1] = *(const uint4*)(p + 16);
  return t.v;
}
// B (32x16 bf16, KxN), weights stored row-major as W[n][k] (i.e. W.T of GEMM):
// lane L<16 = column N=L, K=[0..8) in elems 0-7 and K=[16..24) in elems 8-15; lanes 16-31 same, K shifted by +8.
__device__ __forceinline__ bf16x16 loadB(const __bf16* base, int ldK,
                                         int nBase, int kBase, int lane) {
  int half = lane >> 4;
  int n = nBase + (lane & 15);
  const __bf16* p = base + (size_t)n * ldK + kBase + half * 8;
  ABu t;
  t.q[0] = *(const uint4*)(p);
  t.q[1] = *(const uint4*)(p + 16);
  return t.v;
}

__device__ __forceinline__ f32x8 wmma_bf16(bf16x16 a, bf16x16 b, f32x8 c) {
  return __builtin_amdgcn_wmma_f32_16x16x32_bf16(
      /*neg_a=*/false, a, /*neg_b=*/false, b,
      /*c_mod=*/(short)0, c, /*reuse_a=*/false, /*reuse_b=*/false);
}

__device__ __forceinline__ float sigm(float x) { return 1.0f / (1.0f + __expf(-x)); }

// ===========================================================================
// LSTM kernel (templated: encoder writes lastH at t=T-1, decoder reads h0/c0
// and fuses the per-step H->V logits projection).
// Grid: B/64 blocks x 128 threads (4 waves). Each wave owns 16 batch rows;
// cell state c lives in accumulator-layout VGPRs for all 100 steps.
// ===========================================================================
template <bool DEC>
__global__ __launch_bounds__(128) void lstm_kernel(
    const int* __restrict__ seq,      // (B,T)
    const float* __restrict__ emb,    // (V,E)
    const float* __restrict__ Wih,    // (4H,E)
    const float* __restrict__ Whh,    // (4H,H)
    const float* __restrict__ bias,   // (4H)
    const float* __restrict__ h0,     // (B,H)  [DEC]
    const float* __restrict__ c0,     // (B,H)  [DEC]
    const float* __restrict__ d2vW,   // (V,H)  [DEC]
    const float* __restrict__ d2vb,   // (V)    [DEC]
    float* __restrict__ lastH,        // (B,H)  [ENC out]
    float* __restrict__ logits)       // (B,T-1,V) [DEC out]
{
  // LDS: combined gate weights [Wih | Whh] bf16 (256KB) + emb + h slab + d2v.
  __shared__ __attribute__((aligned(16))) __bf16 Wc[512 * 256];   // W[n][k], k<128:Wih, k>=128:Whh
  __shared__ __attribute__((aligned(16))) __bf16 embL[VV * EE];
  __shared__ __attribute__((aligned(16))) __bf16 hbuf[64 * HH];
  __shared__ __attribute__((aligned(16))) __bf16 d2vL[64 * HH];   // padded V 50->64
  __shared__ float bL[512];
  __shared__ float d2vbL[64];

  const int tid = threadIdx.x;

  for (int idx = tid; idx < 512 * 256; idx += 128) {
    int n = idx >> 8, k = idx & 255;
    float w = (k < 128) ? Wih[n * 128 + k] : Whh[n * 128 + (k - 128)];
    Wc[idx] = (__bf16)w;
  }
  for (int idx = tid; idx < VV * EE; idx += 128) embL[idx] = (__bf16)emb[idx];
  for (int idx = tid; idx < 512; idx += 128) bL[idx] = bias[idx];
  if constexpr (DEC) {
    for (int idx = tid; idx < 64 * 128; idx += 128) {
      int n = idx >> 7, k = idx & 127;
      d2vL[idx] = (__bf16)((n < VV) ? d2vW[n * 128 + k] : 0.0f);
    }
    for (int idx = tid; idx < 64; idx += 128) d2vbL[idx] = (idx < VV) ? d2vb[idx] : 0.0f;
  }

  const int lane  = tid & 31;
  const int wave  = tid >> 5;
  const int half  = lane >> 4;
  const int lane16 = lane & 15;
  const int waveM = wave * 16;             // local row base of this wave
  const int blockRow = blockIdx.x * 64;    // global row base of this WG
  const int mLoc = waveM + lane16;         // this lane's A-row in hbuf
  const int rowA = blockRow + mLoc;        // global batch row for seq/emb

  // Init h slab and cell state.
  if constexpr (DEC) {
    for (int idx = tid; idx < 64 * 128; idx += 128) {
      int m = idx >> 7, k = idx & 127;
      hbuf[idx] = (__bf16)h0[(size_t)(blockRow + m) * HH + k];
    }
  } else {
    for (int idx = tid; idx < 64 * 128; idx += 128) hbuf[idx] = (__bf16)0.0f;
  }
  __syncthreads();

  f32x8 cst[8];  // cell state, accumulator layout: cst[nb][r] <-> (m,n)
  for (int nb = 0; nb < 8; ++nb)
    for (int r = 0; r < 8; ++r) {
      if constexpr (DEC) {
        int m = waveM + r + half * 8;
        int col = nb * 16 + lane16;
        cst[nb][r] = c0[(size_t)(blockRow + m) * HH + col];
      } else {
        cst[nb][r] = 0.0f;
      }
    }

  for (int t = 0; t < TT; ++t) {
    const int tok = seq[(size_t)rowA * TT + t];  // per-lane token id

    // g = [x_t, h] @ [Wih;Whh].T + b  -- 8 column groups x 4 gates, K=256
    for (int nb = 0; nb < 8; ++nb) {
      const int colB = nb * 16;
      f32x8 aI, aF, aG, aO;
      {
        float bi = bL[0 * 128 + colB + lane16];
        float bf = bL[1 * 128 + colB + lane16];
        float bg = bL[2 * 128 + colB + lane16];
        float bo = bL[3 * 128 + colB + lane16];
        for (int r = 0; r < 8; ++r) { aI[r] = bi; aF[r] = bf; aG[r] = bg; aO[r] = bo; }
      }
      for (int kt = 0; kt < 8; ++kt) {
        bf16x16 a = (kt < 4)
            ? loadA_row(embL, EE, tok, kt * 32, half)            // x_t = emb[tok]
            : loadA_row(hbuf, HH, mLoc, (kt - 4) * 32, half);    // h
        aI = wmma_bf16(a, loadB(Wc, 256, 0 * 128 + colB, kt * 32, lane), aI);
        aF = wmma_bf16(a, loadB(Wc, 256, 1 * 128 + colB, kt * 32, lane), aF);
        aG = wmma_bf16(a, loadB(Wc, 256, 2 * 128 + colB, kt * 32, lane), aG);
        aO = wmma_bf16(a, loadB(Wc, 256, 3 * 128 + colB, kt * 32, lane), aO);
      }
      // Gate fusion entirely in registers (same lane/VGPR slot across gates).
      for (int r = 0; r < 8; ++r) {
        float c = sigm(aF[r]) * cst[nb][r] + sigm(aI[r]) * tanhf(aG[r]);
        cst[nb][r] = c;
        float hv = sigm(aO[r]) * tanhf(c);
        int m = waveM + r + half * 8;
        int col = colB + lane16;
        hbuf[(size_t)m * HH + col] = (__bf16)hv;  // own rows only; same-wave LDS is in-order
        if (!DEC && t == TT - 1)
          lastH[(size_t)(blockRow + m) * HH + col] = hv;
      }
    }

    // Decoder: fused logits_t = h_t @ d2v_W.T + d2v_b for t < T-1
    if constexpr (DEC) {
      if (t < TT - 1) {
        for (int nt = 0; nt < 4; ++nt) {
          float bb = d2vbL[nt * 16 + lane16];
          f32x8 acc;
          for (int r = 0; r < 8; ++r) acc[r] = bb;
          for (int kt = 0; kt < 4; ++kt) {
            bf16x16 a = loadA_row(hbuf, HH, mLoc, kt * 32, half);
            acc = wmma_bf16(a, loadB(d2vL, HH, nt * 16, kt * 32, lane), acc);
          }
          int col = nt * 16 + lane16;
          if (col < VV) {
            for (int r = 0; r < 8; ++r) {
              int m = waveM + r + half * 8;
              logits[(size_t)(blockRow + m) * (TT - 1) * VV + (size_t)t * VV + col] = acc[r];
            }
          }
        }
      }
    }
  }
}

// ===========================================================================
// MLP head: sigmoid(W1) -> sigmoid(W2) -> mu/logvar -> z -> h0/c0.
// All six 128x128 weights live in LDS as bf16; per-wave 16-row slabs.
// ===========================================================================
__global__ __launch_bounds__(128) void mlp_kernel(
    const float* __restrict__ lastH,
    const float* __restrict__ W1, const float* __restrict__ b1,
    const float* __restrict__ W2, const float* __restrict__ b2,
    const float* __restrict__ muW, const float* __restrict__ mub,
    const float* __restrict__ lvW, const float* __restrict__ lvb,
    const float* __restrict__ l2hW, const float* __restrict__ l2hb,
    const float* __restrict__ l2cW, const float* __restrict__ l2cb,
    const float* __restrict__ eps,
    float* __restrict__ muOut, float* __restrict__ lvOut,
    float* __restrict__ h0Out, float* __restrict__ c0Out)
{
  __shared__ __attribute__((aligned(16))) __bf16 Wl[6][128 * 128];
  __shared__ float bl[6][128];
  __shared__ __attribute__((aligned(16))) __bf16 buf[64 * 128];

  const int tid = threadIdx.x;
  const float* Ws[6] = {W1, W2, muW, lvW, l2hW, l2cW};
  const float* bs[6] = {b1, b2, mub, lvb, l2hb, l2cb};
  for (int w = 0; w < 6; ++w) {
    for (int idx = tid; idx < 128 * 128; idx += 128) Wl[w][idx] = (__bf16)Ws[w][idx];
    if (tid < 128) bl[w][tid] = bs[w][tid];
  }
  const int blockRow = blockIdx.x * 64;
  for (int idx = tid; idx < 64 * 128; idx += 128) {
    int m = idx >> 7, k = idx & 127;
    buf[idx] = (__bf16)lastH[(size_t)(blockRow + m) * 128 + k];
  }
  __syncthreads();

  const int lane = tid & 31, wave = tid >> 5;
  const int half = lane >> 4, lane16 = lane & 15;
  const int waveM = wave * 16, mLoc = waveM + lane16;

  auto gemm = [&](const __bf16* W, const float* bv, f32x8 acc8[8]) {
    for (int nt = 0; nt < 8; ++nt) {
      float bb = bv[nt * 16 + lane16];
      f32x8 acc;
      for (int r = 0; r < 8; ++r) acc[r] = bb;
      for (int kt = 0; kt < 4; ++kt) {
        bf16x16 a = loadA_row(buf, 128, mLoc, kt * 32, half);
        acc = wmma_bf16(a, loadB(W, 128, nt * 16, kt * 32, lane), acc);
      }
      acc8[nt] = acc;
    }
  };
  auto writeBufSig = [&](f32x8 acc8[8]) {  // activation then store (after all reads)
    for (int nt = 0; nt < 8; ++nt)
      for (int r = 0; r < 8; ++r) {
        int m = waveM + r + half * 8;
        int col = nt * 16 + lane16;
        buf[(size_t)m * 128 + col] = (__bf16)sigm(acc8[nt][r]);
      }
  };

  f32x8 accA[8], accB[8];
  gemm(Wl[0], bl[0], accA);  writeBufSig(accA);   // o = sig(last@W1.T+b1)
  gemm(Wl[1], bl[1], accA);  writeBufSig(accA);   // o = sig(o@W2.T+b2)

  gemm(Wl[2], bl[2], accA);                       // mu
  gemm(Wl[3], bl[3], accB);                       // logvar
  for (int nt = 0; nt < 8; ++nt)
    for (int r = 0; r < 8; ++r) {
      int m = waveM + r + half * 8;
      int col = nt * 16 + lane16;
      size_t gi = (size_t)(blockRow + m) * 128 + col;
      float mu = accA[nt][r], lv = accB[nt][r];
      muOut[gi] = mu;
      lvOut[gi] = lv;
      float z = mu + __expf(0.5f * lv) * eps[gi];
      buf[(size_t)m * 128 + col] = (__bf16)z;
    }

  gemm(Wl[4], bl[4], accA);                       // h0 = z@l2h.T+b
  gemm(Wl[5], bl[5], accB);                       // c0 = z@l2c.T+b
  for (int nt = 0; nt < 8; ++nt)
    for (int r = 0; r < 8; ++r) {
      int m = waveM + r + half * 8;
      int col = nt * 16 + lane16;
      size_t gi = (size_t)(blockRow + m) * 128 + col;
      h0Out[gi] = accA[nt][r];
      c0Out[gi] = accB[nt][r];
    }
}

// ===========================================================================
extern "C" void kernel_launch(void* const* d_in, const int* in_sizes, int n_in,
                              void* d_out, int out_size, void* d_ws, size_t ws_size,
                              hipStream_t stream) {
  const int*   in_seq  = (const int*)d_in[0];
  const int*   out_seq = (const int*)d_in[1];
  const float* eps     = (const float*)d_in[2];
  const float* emb     = (const float*)d_in[3];
  const float* enc_Wih = (const float*)d_in[4];
  const float* enc_Whh = (const float*)d_in[5];
  const float* enc_b   = (const float*)d_in[6];
  const float* e2o_W1  = (const float*)d_in[7];
  const float* e2o_b1  = (const float*)d_in[8];
  const float* e2o_W2  = (const float*)d_in[9];
  const float* e2o_b2  = (const float*)d_in[10];
  const float* mu_W    = (const float*)d_in[11];
  const float* mu_b    = (const float*)d_in[12];
  const float* lv_W    = (const float*)d_in[13];
  const float* lv_b    = (const float*)d_in[14];
  const float* l2h_W   = (const float*)d_in[15];
  const float* l2h_b   = (const float*)d_in[16];
  const float* l2c_W   = (const float*)d_in[17];
  const float* l2c_b   = (const float*)d_in[18];
  const float* dec_Wih = (const float*)d_in[19];
  const float* dec_Whh = (const float*)d_in[20];
  const float* dec_b   = (const float*)d_in[21];
  const float* d2v_W   = (const float*)d_in[22];
  const float* d2v_b   = (const float*)d_in[23];

  float* out    = (float*)d_out;
  float* logits = out;                                      // (B, T-1, V)
  float* muOut  = out + (size_t)BB * (TT - 1) * VV;          // (B, L)
  float* lvOut  = muOut + (size_t)BB * LL_;                  // (B, L)

  float* ws    = (float*)d_ws;
  float* lastH = ws;                       // (B,H)
  float* h0    = ws + (size_t)BB * HH;     // (B,H)
  float* c0    = h0 + (size_t)BB * HH;     // (B,H)

  dim3 grid(BB / 64), block(128);
  lstm_kernel<false><<<grid, block, 0, stream>>>(
      in_seq, emb, enc_Wih, enc_Whh, enc_b,
      nullptr, nullptr, nullptr, nullptr, lastH, nullptr);

  mlp_kernel<<<grid, block, 0, stream>>>(
      lastH, e2o_W1, e2o_b1, e2o_W2, e2o_b2, mu_W, mu_b, lv_W, lv_b,
      l2h_W, l2h_b, l2c_W, l2c_b, eps, muOut, lvOut, h0, c0);

  lstm_kernel<true><<<grid, block, 0, stream>>>(
      out_seq, emb, dec_Wih, dec_Whh, dec_b,
      h0, c0, d2v_W, d2v_b, nullptr, logits);
}
